// Mamba_40192303956452
// MI455X (gfx1250) — matrix operand order repaired and checked by reading, not compiled, not run on hardware
//
#include <hip/hip_runtime.h>
#include <hip/hip_bf16.h>
#include <math.h>

// ---------------- problem constants ----------------
#define B_ 2
#define L_ 2048
#define E_ 1184
#define D_ 4048
#define N_ 64
#define DR_ 64
#define KCONV_ 4
#define TWO_D_ (2 * D_)
#define M_ (B_ * L_)        // 4096 tokens
#define XR_ (DR_ + 2 * N_)  // 192

typedef __attribute__((ext_vector_type(16))) __bf16 v16bf;
typedef __attribute__((ext_vector_type(8)))  __bf16 v8bf;
typedef __attribute__((ext_vector_type(8)))  float  v8f;
typedef __attribute__((ext_vector_type(4)))  int    v4i;

#if __has_builtin(__builtin_amdgcn_global_load_async_to_lds_b128)
#define USE_ASYNC 1
#else
#define USE_ASYNC 0
#endif

#if USE_ASYNC
typedef __attribute__((address_space(1))) v4i g_v4i;
typedef __attribute__((address_space(3))) v4i l_v4i;
__device__ __forceinline__ void async_cp16(const void* g, void* l) {
    __builtin_amdgcn_global_load_async_to_lds_b128((g_v4i*)g, (l_v4i*)l, 0, 0);
}
template <int Ncnt>
__device__ __forceinline__ void wait_async() {
#if __has_builtin(__builtin_amdgcn_s_wait_asynccnt)
    __builtin_amdgcn_s_wait_asynccnt(Ncnt);
#else
    asm volatile("s_wait_asynccnt %0" ::"n"(Ncnt) : "memory");
#endif
}
#else
template <int Ncnt>
__device__ __forceinline__ void wait_async() {}
#endif

__device__ __forceinline__ unsigned short f2bf_bits(float f) {
    union { float f; unsigned int u; } c; c.f = f;
    unsigned int r = c.u + 0x7FFFu + ((c.u >> 16) & 1u);  // RNE
    return (unsigned short)(r >> 16);
}
__device__ __forceinline__ float silu_f(float x) {
    return x * __builtin_amdgcn_rcpf(1.0f + __expf(-x));
}

// =====================================================================
// WMMA bf16 GEMM:  C[M,N] = act( A[M,K]_bf16 * W[N,K]_bf16^T + bias[N] )
// 64x128 block tile, 8 waves (2x4) of 32x32; K-tiles of 32.
// Tiles staged via double-buffered GLOBAL_LOAD_ASYNC_TO_LDS_B128:
// each thread owns 3 16B chunks/tile (1 A + 2 W) -> ASYNCcnt uniform = 3.
// Requires: lda/ldw multiples of 8, K multiple of 8, base ptrs 16B aligned.
// =====================================================================
__global__ __launch_bounds__(256)
void gemm_bf16_wmma(const unsigned short* __restrict__ A, int lda,
                    const unsigned short* __restrict__ W, int ldw,
                    const float* __restrict__ bias,
                    float* __restrict__ C, int ldc,
                    int M, int N, int K, int act) {
    __shared__ __align__(16) __bf16 As[2][64][40];   // +8 pad rotates banks
    __shared__ __align__(16) __bf16 Ws[2][128][40];
    __shared__ __align__(16) __bf16 dump[16];        // sink for OOB async lanes

    const int tid  = threadIdx.x;
    const int wave = tid >> 5, lane = tid & 31;
    const int wm = wave >> 2, wn = wave & 3;
    const int m0 = blockIdx.y * 64, n0 = blockIdx.x * 128;

    const int rA = tid >> 2, cA = (tid & 3) * 8;   // A: 64x32 = 256 chunks
    const int rW = tid >> 1, cW = (tid & 1) * 16;  // W: 128x32 = 512 chunks
    const bool mok = (m0 + rA) < M;
    const bool nok = (n0 + rW) < N;

    auto issue = [&](int kk, int s) {
        const int kb = kk << 5;
        {   // A chunk
            const bool ok = mok && (kb + cA + 8 <= K);
            const unsigned short* g = ok ? A + (size_t)(m0 + rA) * lda + kb + cA : A;
            __bf16* l = &As[s][rA][cA];
#if USE_ASYNC
            async_cp16(g, ok ? l : (__bf16*)dump);
            if (!ok) { v8bf z = {}; *(v8bf*)l = z; }
#else
            if (ok) *(uint4*)l = *(const uint4*)g;
            else    { v8bf z = {}; *(v8bf*)l = z; }
#endif
        }
        #pragma unroll
        for (int h = 0; h < 2; ++h) {  // two W chunks
            const int c = cW + h * 8;
            const bool ok = nok && (kb + c + 8 <= K);
            const unsigned short* g = ok ? W + (size_t)(n0 + rW) * ldw + kb + c : W;
            __bf16* l = &Ws[s][rW][c];
#if USE_ASYNC
            async_cp16(g, ok ? l : (__bf16*)dump);
            if (!ok) { v8bf z = {}; *(v8bf*)l = z; }
#else
            if (ok) *(uint4*)l = *(const uint4*)g;
            else    { v8bf z = {}; *(v8bf*)l = z; }
#endif
        }
    };

    v8f acc[2][2] = {};
    const int kt = (K + 31) >> 5;
    issue(0, 0);
    for (int kk = 0; kk < kt; ++kk) {
        const int s = kk & 1;
        if (kk + 1 < kt) { issue(kk + 1, s ^ 1); wait_async<3>(); }
        else             { wait_async<0>(); }
        __syncthreads();

        // A frags: lanes 0-15 K{0-7,16-23}, lanes 16-31 K{8-15,24-31}
        const int ar = lane & 15, akb = (lane < 16) ? 0 : 8;
        v16bf afrag[2], bfrag[2];
        #pragma unroll
        for (int i = 0; i < 2; ++i) {
            const v8bf lo = *(const v8bf*)&As[s][wm * 32 + i * 16 + ar][akb];
            const v8bf hi = *(const v8bf*)&As[s][wm * 32 + i * 16 + ar][akb + 16];
            #pragma unroll
            for (int e = 0; e < 8; ++e) { afrag[i][e] = lo[e]; afrag[i][e + 8] = hi[e]; }
        }
        // B frags: lane col = lane&15; lanes 0-15 K{0-7,16-23}, lanes 16-31 K{8-15,24-31}
        const int bn = lane & 15, bkb = (lane < 16) ? 0 : 8;
        #pragma unroll
        for (int j = 0; j < 2; ++j) {
            const v8bf lo = *(const v8bf*)&Ws[s][wn * 32 + j * 16 + bn][bkb];
            const v8bf hi = *(const v8bf*)&Ws[s][wn * 32 + j * 16 + bn][bkb + 16];
            #pragma unroll
            for (int e = 0; e < 8; ++e) { bfrag[j][e] = lo[e]; bfrag[j][e + 8] = hi[e]; }
        }
        #pragma unroll
        for (int i = 0; i < 2; ++i)
            #pragma unroll
            for (int j = 0; j < 2; ++j)
                acc[i][j] = __builtin_amdgcn_wmma_f32_16x16x32_bf16(
                    false, afrag[i], false, bfrag[j],
                    (short)0, acc[i][j], false, false);
        __syncthreads();
    }

    // epilogue: VGPR v -> row (v + 8*(lane>=16)), col = lane&15
    const int rl = lane & 15, rh = (lane >> 4) * 8;
    #pragma unroll
    for (int i = 0; i < 2; ++i) {
        #pragma unroll
        for (int j = 0; j < 2; ++j) {
            const int col = n0 + wn * 32 + j * 16 + rl;
            if (col >= N) continue;
            const float bz = bias ? bias[col] : 0.0f;
            #pragma unroll
            for (int v = 0; v < 8; ++v) {
                const int row = m0 + wm * 32 + i * 16 + rh + v;
                if (row >= M) continue;
                float o = acc[i][j][v] + bz;
                if (act == 1) o = (o > 20.0f) ? o : log1pf(__expf(o));  // softplus
                C[(size_t)row * ldc + col] = o;
            }
        }
    }
}

// =====================================================================
// f32 -> bf16 conversion, 2D strided
// =====================================================================
__global__ __launch_bounds__(256)
void cvt_bf16_kernel(const float* __restrict__ src, int sld,
                     unsigned short* __restrict__ dst, int dld,
                     int rows, int cols) {
    const size_t idx = (size_t)blockIdx.x * 256 + threadIdx.x;
    if (idx >= (size_t)rows * cols) return;
    const int c = (int)(idx % cols);
    const size_t r = idx / cols;
    dst[r * dld + c] = f2bf_bits(src[r * sld + c]);
}

// =====================================================================
// Depthwise causal conv (K=4) + SiLU; emits f32 (scan) and bf16 (GEMM).
// =====================================================================
__global__ __launch_bounds__(256)
void conv_silu_kernel(const float* __restrict__ xz,
                      const float* __restrict__ cw,
                      const float* __restrict__ cb,
                      float* __restrict__ xc_f,
                      unsigned short* __restrict__ xc_h) {
    const size_t idx = (size_t)blockIdx.x * 256 + threadIdx.x;
    if (idx >= (size_t)M_ * D_) return;
    const int d = (int)(idx % D_);
    const size_t tok = idx / D_;
    const int t = (int)(tok % L_);
    const size_t brow = tok - t;  // b*L
    float acc = cb[d];
    #pragma unroll
    for (int k = 0; k < KCONV_; ++k) {
        const int ts = t - (KCONV_ - 1) + k;
        if (ts >= 0)
            acc += cw[d * KCONV_ + k] * xz[(brow + ts) * TWO_D_ + d];
    }
    const float s = silu_f(acc);
    xc_f[idx] = s;
    xc_h[idx] = f2bf_bits(s);
}

// =====================================================================
// yg = bf16( y * silu(z) ),  z = xz[:, D:2D]
// =====================================================================
__global__ __launch_bounds__(256)
void gate_kernel(const float* __restrict__ y,
                 const float* __restrict__ xz,
                 unsigned short* __restrict__ yg) {
    const size_t idx = (size_t)blockIdx.x * 256 + threadIdx.x;
    if (idx >= (size_t)M_ * D_) return;
    const int d = (int)(idx % D_);
    const size_t tok = idx / D_;
    const float g = xz[tok * TWO_D_ + D_ + d];
    yg[idx] = f2bf_bits(y[idx] * silu_f(g));
}

// =====================================================================
// Selective scan: wave per (b,d), lane holds states n=lane, lane+32.
// B/C rows + per-d delta/u staged in 64-step LDS chunks per block.
// =====================================================================
#define TCH 64
__global__ __launch_bounds__(256)
void scan_kernel(const float* __restrict__ xr,     // [M,192] : [dr | B | C]
                 const float* __restrict__ delta,  // [M,D]
                 const float* __restrict__ xc,     // [M,D] (u)
                 const float* __restrict__ A_log,  // [D,N]
                 const float* __restrict__ Dp,     // [D]
                 float* __restrict__ y) {          // [M,D]
    __shared__ float Bs[TCH][64];
    __shared__ float Cs[TCH][64];
    __shared__ float Dls[TCH][8];
    __shared__ float Uls[TCH][8];

    const int tid = threadIdx.x, wave = tid >> 5, lane = tid & 31;
    const int b = blockIdx.y;
    const int d = blockIdx.x * 8 + wave;
    const size_t tokb = (size_t)b * L_;

    const float A0 = -__expf(A_log[(size_t)d * N_ + lane]);
    const float A1 = -__expf(A_log[(size_t)d * N_ + lane + 32]);
    const float dpd = Dp[d];
    float h0 = 0.0f, h1 = 0.0f;

    for (int t0 = 0; t0 < L_; t0 += TCH) {
        for (int e = tid; e < TCH * 64; e += 256) {
            const int tt = e >> 6, n = e & 63;
            const float* row = xr + (tokb + t0 + tt) * XR_;
            Bs[tt][n] = row[DR_ + n];
            Cs[tt][n] = row[DR_ + N_ + n];
        }
        for (int e = tid; e < TCH * 8; e += 256) {
            const int tt = e >> 3, wd = e & 7;
            const size_t off = (tokb + t0 + tt) * D_ + (size_t)blockIdx.x * 8 + wd;
            Dls[tt][wd] = delta[off];
            Uls[tt][wd] = xc[off];
        }
        __syncthreads();

        for (int tt = 0; tt < TCH; ++tt) {
            const float dt = Dls[tt][wave];
            const float u  = Uls[tt][wave];
            const float b0 = Bs[tt][lane], b1 = Bs[tt][lane + 32];
            const float c0 = Cs[tt][lane], c1 = Cs[tt][lane + 32];
            h0 = __expf(dt * A0) * h0 + dt * b0 * u;
            h1 = __expf(dt * A1) * h1 + dt * b1 * u;
            float yv = h0 * c0 + h1 * c1;
            #pragma unroll
            for (int off = 16; off; off >>= 1)
                yv += __shfl_xor(yv, off, 32);
            if (lane == 0)
                y[(tokb + t0 + tt) * D_ + d] = yv + u * dpd;
        }
        __syncthreads();
    }
}

// =====================================================================
extern "C" void kernel_launch(void* const* d_in, const int* in_sizes, int n_in,
                              void* d_out, int out_size, void* d_ws, size_t ws_size,
                              hipStream_t stream) {
    const float* x       = (const float*)d_in[0];
    const float* in_w    = (const float*)d_in[1];
    const float* in_b    = (const float*)d_in[2];
    const float* conv_w  = (const float*)d_in[3];
    const float* conv_b  = (const float*)d_in[4];
    const float* xproj_w = (const float*)d_in[5];
    const float* xproj_b = (const float*)d_in[6];
    const float* dproj_w = (const float*)d_in[7];
    const float* dproj_b = (const float*)d_in[8];
    const float* A_log   = (const float*)d_in[9];
    const float* Dp      = (const float*)d_in[10];
    const float* out_w   = (const float*)d_in[11];
    const float* out_b   = (const float*)d_in[12];
    float* out = (float*)d_out;

    char* ws = (char*)d_ws;
    size_t off = 0;
    auto alloc = [&](size_t bytes) {
        void* p = ws + off;
        off = (off + bytes + 255) & ~(size_t)255;
        return p;
    };
    float* xz    = (float*)alloc((size_t)M_ * TWO_D_ * 4);
    float* xc_f  = (float*)alloc((size_t)M_ * D_ * 4);
    float* xr    = (float*)alloc((size_t)M_ * XR_ * 4);
    float* delta = (float*)alloc((size_t)M_ * D_ * 4);
    float* yscan = (float*)alloc((size_t)M_ * D_ * 4);
    unsigned short* x_h  = (unsigned short*)alloc((size_t)M_ * E_ * 2);
    unsigned short* iw_h = (unsigned short*)alloc((size_t)TWO_D_ * E_ * 2);
    unsigned short* xw_h = (unsigned short*)alloc((size_t)XR_ * D_ * 2);
    unsigned short* dw_h = (unsigned short*)alloc((size_t)D_ * DR_ * 2);
    unsigned short* ow_h = (unsigned short*)alloc((size_t)E_ * D_ * 2);
    unsigned short* xc_h = (unsigned short*)alloc((size_t)M_ * D_ * 2);
    unsigned short* dr_h = (unsigned short*)alloc((size_t)M_ * DR_ * 2);
    unsigned short* yg_h = (unsigned short*)alloc((size_t)M_ * D_ * 2);
    (void)ws_size; (void)in_sizes; (void)n_in; (void)out_size;

    auto cvt = [&](const float* s, int sld, unsigned short* d, int dld,
                   int rows, int cols) {
        const size_t tot = (size_t)rows * cols;
        cvt_bf16_kernel<<<(unsigned)((tot + 255) / 256), 256, 0, stream>>>(
            s, sld, d, dld, rows, cols);
    };

    // 0) pre-convert GEMM operands to bf16
    cvt(x,       E_, x_h,  E_, M_,     E_);
    cvt(in_w,    E_, iw_h, E_, TWO_D_, E_);
    cvt(xproj_w, D_, xw_h, D_, XR_,    D_);
    cvt(dproj_w, DR_, dw_h, DR_, D_,   DR_);
    cvt(out_w,   D_, ow_h, D_, E_,     D_);

    // 1) xz = x @ in_w^T + in_b          [4096, 8096]
    gemm_bf16_wmma<<<dim3((TWO_D_ + 127) / 128, M_ / 64), 256, 0, stream>>>(
        x_h, E_, iw_h, E_, in_b, xz, TWO_D_, M_, TWO_D_, E_, 0);

    // 2) xc = silu(conv(xz[:, :D]))      [4096, 4048]  (f32 + bf16)
    conv_silu_kernel<<<(unsigned)(((size_t)M_ * D_ + 255) / 256), 256, 0, stream>>>(
        xz, conv_w, conv_b, xc_f, xc_h);

    // 3) xr = xc @ xproj_w^T + xproj_b   [4096, 192]
    gemm_bf16_wmma<<<dim3((XR_ + 127) / 128, M_ / 64), 256, 0, stream>>>(
        xc_h, D_, xw_h, D_, xproj_b, xr, XR_, M_, XR_, D_, 0);

    // 4) delta = softplus(xr[:, :64] @ dproj_w^T + dproj_b)  [4096, 4048]
    cvt(xr, XR_, dr_h, DR_, M_, DR_);
    gemm_bf16_wmma<<<dim3((D_ + 127) / 128, M_ / 64), 256, 0, stream>>>(
        dr_h, DR_, dw_h, DR_, dproj_b, delta, D_, M_, D_, DR_, 1);

    // 5) selective scan -> yscan [4096, 4048] (includes +u*Dp)
    scan_kernel<<<dim3(D_ / 8, B_), 256, 0, stream>>>(
        xr, delta, xc_f, A_log, Dp, yscan);

    // 6) yg = bf16(yscan * silu(z));  out = yg @ out_w^T + out_b  [4096, 1184]
    gate_kernel<<<(unsigned)(((size_t)M_ * D_ + 255) / 256), 256, 0, stream>>>(
        yscan, xz, yg_h);
    gemm_bf16_wmma<<<dim3((E_ + 127) / 128, M_ / 64), 256, 0, stream>>>(
        yg_h, D_, ow_h, D_, out_b, out, E_, M_, E_, D_, 0);
}
